// FourierLayer2d_7009386627311
// MI455X (gfx1250) — matrix operand order repaired and checked by reading, not compiled, not run on hardware
//
#include <hip/hip_runtime.h>
#include <hip/hip_bf16.h>

typedef __attribute__((ext_vector_type(16))) _Float16 v16h;
typedef __attribute__((ext_vector_type(8)))  _Float16 v8h;
typedef __attribute__((ext_vector_type(8)))  float    v8f;

union V16H { v16h v; v8h h[2]; };

#define WMMA_F16(a, b, c) \
  __builtin_amdgcn_wmma_f32_16x16x32_f16(false, (a), false, (b), (short)0, (c), false, false)

// A-matrix (16x32 f16) lane load: halves 0-7 = K kbase+g*8.., halves 8-15 = K kbase+16+g*8..
__device__ __forceinline__ v16h load_a16(const _Float16* base, int row, int stride, int kbase, int g) {
  V16H r;
  const _Float16* p = base + (size_t)row * stride + kbase;
  r.h[0] = *(const v8h*)(p + g * 8);
  r.h[1] = *(const v8h*)(p + 16 + g * 8);
  return r.v;
}
// B-matrix (32x16 f16) lane load from TRANSPOSED [col][k] buffer: halves 0-7 = K kbase+g*8.., halves 8-15 = K kbase+16+g*8..
__device__ __forceinline__ v16h load_bT16(const _Float16* base, int col, int stride, int kbase, int g) {
  V16H r;
  const _Float16* p = base + (size_t)col * stride + kbase;
  r.h[0] = *(const v8h*)(p + g * 8);
  r.h[1] = *(const v8h*)(p + 16 + g * 8);
  return r.v;
}

// ---------------- constants ----------------
// B=8, M=256, N=256, I=O=64, NM=16; modes: ky(j) = j<16 ? j : 224+j  (240..255)

// ---------------- workspace layout (bytes) ----------------
#define OFF_TA   ((size_t)0)              // [32][256] f16  (stage A twiddle)
#define OFF_A2   ((size_t)16384)          // [64][512] f16  (stage B twiddle [C|S]/[-S|C])
#define OFF_A4   ((size_t)81920)          // [512][64] f16  (stage D phase, 1/256 folded)
#define OFF_A5   ((size_t)147456)         // [256][32] f16  (stage E phase, c_k & 1/256 folded)
#define OFF_LWT  ((size_t)163840)         // [64][64]  f16  (lin_w transposed)
#define OFF_WBT  ((size_t)172032)         // [32][16][2][64][128] f16 (packed spectral weights, transposed)
#define OFF_X1   ((size_t)16949248)       // [8][16][2][256][64] f16
#define OFF_X2   ((size_t)25337856)       // [16(pad)][16][64][64] f16
#define OFF_OFT  ((size_t)27435008)       // [8][16][64][64] f16
#define OFF_Y1   ((size_t)28483584)       // [8][256][2][16][64] f16
// total ~36.9 MB

// ---------------- init: small DFT/phase tables ----------------
__global__ __launch_bounds__(256) void k_tables(_Float16* TA, _Float16* A2, _Float16* A4,
                                                _Float16* A5, _Float16* LWT, const float* lin_w) {
  int tid = blockIdx.x * 256 + threadIdx.x;   // 32768 threads
  const float TP = 6.28318530717958647692f;
  if (tid < 8192) {                            // TA [32][256]: rows 0-15 cos, 16-31 -sin
    int r = tid >> 8, n = tid & 255;
    int kx = r & 15;
    float th = TP * (float)((kx * n) & 255) / 256.0f;
    TA[tid] = (_Float16)((r < 16) ? cosf(th) : -sinf(th));
  }
  {                                            // A2 [64][512]: rows 0-31 [C|S], 32-63 [-S|C]
    int r = tid >> 9, k = tid & 511;
    int j = r & 31; int ky = (j < 16) ? j : (224 + j);
    int m = k & 255;
    float th = TP * (float)((ky * m) & 255) / 256.0f;
    float c = cosf(th), s = sinf(th);
    float v = (r < 32) ? ((k < 256) ? c : s) : ((k < 256) ? -s : c);
    A2[tid] = (_Float16)v;
  }
  {                                            // A4 [512][64]: rows 0-255 [C|-S]/256, 256-511 [S|C]/256
    int r = tid >> 6, k = tid & 63;
    int ri = r >> 8, m = r & 255;
    int kk = k & 31; int ky = (kk < 16) ? kk : (224 + kk);
    float th = TP * (float)((ky * m) & 255) / 256.0f;
    float c = cosf(th), s = sinf(th);
    float v = (ri == 0) ? ((k < 32) ? c : -s) : ((k < 32) ? s : c);
    A4[tid] = (_Float16)(v * (1.0f / 256.0f));
  }
  if (tid < 8192) {                            // A5 [256][32]: [c_k cos | -c_k sin]/256
    int n = tid >> 5, k = tid & 31;
    int kx = k & 15;
    float th = TP * (float)((kx * n) & 255) / 256.0f;
    float ck = (kx == 0) ? 1.0f : 2.0f;
    float v = (k < 16) ? ck * cosf(th) : -ck * sinf(th);
    A5[tid] = (_Float16)(v * (1.0f / 256.0f));
  }
  if (tid < 4096) {                            // LWT [o2][o] = lin_w[o][o2]
    int o2 = tid >> 6, o = tid & 63;
    LWT[tid] = (_Float16)lin_w[o * 64 + o2];
  }
}

// init: spectral weights packed as WBT[j][y][ri][o][k]  (B transposed: [col o][K])
// ri=0 -> [Wr ; -Wi] (for out_r), ri=1 -> [Wi ; Wr] (for out_i); K = 2*I = 128 (i then i again)
__global__ __launch_bounds__(256) void k_wbt(const float* w1, const float* w2, _Float16* WBT) {
  size_t idx = (size_t)blockIdx.x * 256 + threadIdx.x;   // 8,388,608 total
  int k  = (int)(idx & 127);
  int o  = (int)((idx >> 7) & 63);
  int ri = (int)((idx >> 13) & 1);
  int y  = (int)((idx >> 14) & 15);
  int j  = (int)(idx >> 18);
  const float* w = (j < 16) ? w1 : w2;
  int x = j & 15;
  int i = k & 63;
  int comp = ri ^ (k >> 6);
  float sgn = (ri == 0 && k >= 64) ? -1.0f : 1.0f;
  float val = sgn * w[((((size_t)i * 64 + o) * 16 + x) * 16 + y) * 2 + comp];
  WBT[idx] = (_Float16)val;
}

// ---------------- Stage A: column DFT over n -> X1[b][kx][ri][m][i] ----------------
__global__ __launch_bounds__(256) void k_stageA(const float* __restrict__ x,
                                                const _Float16* __restrict__ TA,
                                                _Float16* __restrict__ X1) {
  __shared__ __align__(16) _Float16 BTx[64][256];   // x-row transposed [i][n], f16, 32 KB
  int bm = blockIdx.x;                 // b*256 + m
  int b = bm >> 8, m = bm & 255;
  int tid = threadIdx.x;
  const float* xp = x + (size_t)bm * (256 * 64);
  for (int e = tid; e < 256 * 64; e += 256) {
    int n = e >> 6, i = e & 63;
    BTx[i][n] = (_Float16)xp[e];
  }
  __syncthreads();
  int w = tid >> 5, lane = tid & 31, g = lane >> 4, mr = lane & 15;
  int rtile = w >> 2, itile = w & 3;   // 2 row-tiles (r;i) x 4 i-tiles
  v8f acc = {};
  for (int k0 = 0; k0 < 256; k0 += 32) {
    v16h a  = load_a16(TA, rtile * 16 + mr, 256, k0, g);
    v16h bb = load_bT16(&BTx[0][0], itile * 16 + mr, 256, k0, g);
    acc = WMMA_F16(a, bb, acc);
  }
  int i = itile * 16 + mr;
  for (int v = 0; v < 8; ++v) {
    int r = rtile * 16 + v + 8 * g;    // 0..31: kx = r%16, ri = r/16
    int kx = r & 15, ri = r >> 4;
    X1[((((size_t)b * 16 + kx) * 2 + ri) * 256 + m) * 64 + i] = (_Float16)acc[v];
  }
}

// ---------------- Stage B: row DFT over m -> X2[b][kx][64 rows][i] ----------------
__global__ __launch_bounds__(256) void k_stageB(const _Float16* __restrict__ X1,
                                                const _Float16* __restrict__ A2,
                                                _Float16* __restrict__ X2) {
  __shared__ __align__(16) _Float16 BT1[64][256];   // half of [512 k][64 i], transposed; 32 KB
  int bk = blockIdx.x;                 // b*16 + kx
  int b = bk >> 4, kx = bk & 15;
  int tid = threadIdx.x;
  int w = tid >> 5, lane = tid & 31, g = lane >> 4, mr = lane & 15;
  v8f acc[2] = {{}, {}};
  for (int half = 0; half < 2; ++half) {
    __syncthreads();
    const _Float16* xp = X1 + (size_t)bk * (512 * 64) + (size_t)half * (256 * 64);
    for (int e = tid; e < 256 * 64; e += 256) {
      int k = e >> 6, i = e & 63;
      BT1[i][k] = xp[e];
    }
    __syncthreads();
    for (int s = 0; s < 2; ++s) {
      int t = w * 2 + s;
      int rtile = t >> 2, itile = t & 3;
      for (int k0 = 0; k0 < 256; k0 += 32) {
        v16h a  = load_a16(A2, rtile * 16 + mr, 512, half * 256 + k0, g);
        v16h bb = load_bT16(&BT1[0][0], itile * 16 + mr, 256, k0, g);
        acc[s] = WMMA_F16(a, bb, acc[s]);
      }
    }
  }
  for (int s = 0; s < 2; ++s) {
    int t = w * 2 + s;
    int rtile = t >> 2, itile = t & 3;
    int i = itile * 16 + mr;
    for (int v = 0; v < 8; ++v) {
      int r = rtile * 16 + v + 8 * g;  // 0..31 = X2r(mode), 32..63 = X2i(mode)
      X2[(((size_t)b * 16 + kx) * 64 + r) * 64 + i] = (_Float16)acc[s][v];
    }
  }
}

// ---------------- Stage C: per-mode spectral mix -> OFT[b][kx][ri*32+j][o] ----------------
__global__ __launch_bounds__(256) void k_stageC(const _Float16* __restrict__ X2,
                                                const _Float16* __restrict__ WBT,
                                                _Float16* __restrict__ OFT) {
  int jk = blockIdx.x;                 // j*16 + kx
  int j = jk >> 4, kx = jk & 15;
  int tid = threadIdx.x, w = tid >> 5, lane = tid & 31, g = lane >> 4, mr = lane & 15;
  int ri = w >> 2, otile = w & 3;      // 2 (out r/i) x 4 o-tiles = 8 waves
  const _Float16* wb = WBT + ((((size_t)j * 16 + kx) * 2 + ri) * 64) * 128;
  v8f acc = {};
  for (int k0 = 0; k0 < 128; k0 += 32) {
    // A rows = batch b (mr; rows 8-15 read padded garbage, never stored)
    int K1 = k0 + g * 8, K2 = k0 + 16 + g * 8;
    const _Float16* x2b = X2 + (((size_t)mr * 16 + kx) * 64) * 64;
    const _Float16* p1 = x2b + (size_t)((K1 < 64) ? j : (32 + j)) * 64 + (K1 & 63);
    const _Float16* p2 = x2b + (size_t)((K2 < 64) ? j : (32 + j)) * 64 + (K2 & 63);
    V16H a;
    a.h[0] = *(const v8h*)p1;
    a.h[1] = *(const v8h*)p2;
    v16h bb = load_bT16(wb, otile * 16 + mr, 128, k0, g);
    acc = WMMA_F16(a.v, bb, acc);
  }
  int o = otile * 16 + mr;
  for (int v = 0; v < 8; ++v) {
    int b = v + 8 * g;
    if (b < 8)
      OFT[(((size_t)b * 16 + kx) * 64 + ri * 32 + j) * 64 + o] = (_Float16)acc[v];
  }
}

// ---------------- Stage D: inverse row DFT -> Y1[b][m][ri][kx][o] ----------------
__global__ __launch_bounds__(256) void k_stageD(const _Float16* __restrict__ OFT,
                                                const _Float16* __restrict__ A4,
                                                _Float16* __restrict__ Y1) {
  __shared__ __align__(16) _Float16 BT4[64][64];    // [o][k], 8 KB
  int bk = blockIdx.x;                 // b*16 + kx
  int b = bk >> 4, kx = bk & 15;
  int tid = threadIdx.x;
  const _Float16* op = OFT + (size_t)bk * (64 * 64);
  for (int e = tid; e < 64 * 64; e += 256) {
    int k = e >> 6, o = e & 63;
    BT4[o][k] = op[e];
  }
  __syncthreads();
  int w = tid >> 5, lane = tid & 31, g = lane >> 4, mr = lane & 15;
  for (int t = w * 16; t < w * 16 + 16; ++t) {
    int rtile = t >> 2, otile = t & 3;
    v8f acc = {};
    for (int k0 = 0; k0 < 64; k0 += 32) {
      v16h a  = load_a16(A4, rtile * 16 + mr, 64, k0, g);
      v16h bb = load_bT16(&BT4[0][0], otile * 16 + mr, 64, k0, g);
      acc = WMMA_F16(a, bb, acc);
    }
    int o = otile * 16 + mr;
    for (int v = 0; v < 8; ++v) {
      int r = rtile * 16 + v + 8 * g;  // 0..511: ri = r/256, m = r%256
      int riY = r >> 8, m = r & 255;
      Y1[((((size_t)b * 256 + m) * 2 + riY) * 16 + kx) * 64 + o] = (_Float16)acc[v];
    }
  }
}

// ---------------- Stage E: inverse rfft over n (fused) + linear + bias + ReLU ----------------
__global__ __launch_bounds__(256) void k_stageE(const _Float16* __restrict__ Y1,
                                                const _Float16* __restrict__ A5,
                                                const _Float16* __restrict__ LWT,
                                                const float* __restrict__ lin_b,
                                                float* __restrict__ out) {
  __shared__ __align__(16) _Float16 BT5[64][32];    // [o][k'], 4 KB
  __shared__ __align__(16) _Float16 yL[256][72];    // y [n][o] f16, padded stride; 36 KB
  int bm = blockIdx.x;                 // b*256 + m
  int tid = threadIdx.x;
  const _Float16* yp = Y1 + (size_t)bm * (32 * 64);
  for (int e = tid; e < 32 * 64; e += 256) {
    int k = e >> 6, o = e & 63;
    BT5[o][k] = yp[e];
  }
  __syncthreads();
  int w = tid >> 5, lane = tid & 31, g = lane >> 4, mr = lane & 15;
  v8f yt[8];
  for (int s = 0; s < 8; ++s) {        // phase 1: y = A5 x [Y1r;Y1i], K=32 (one WMMA step)
    int t = w * 8 + s;
    int ntile = t >> 2, otile = t & 3;
    v8f acc = {};
    v16h a  = load_a16(A5, ntile * 16 + mr, 32, 0, g);
    v16h bb = load_bT16(&BT5[0][0], otile * 16 + mr, 32, 0, g);
    acc = WMMA_F16(a, bb, acc);
    yt[s] = acc;
    int o = otile * 16 + mr;
    for (int v = 0; v < 8; ++v) {
      int n = ntile * 16 + v + 8 * g;
      yL[n][o] = (_Float16)acc[v];
    }
  }
  __syncthreads();
  float* op = out + (size_t)bm * (256 * 64);
  for (int s = 0; s < 8; ++s) {        // phase 2: out = relu(y + y*W + bias), K=64
    int t = w * 8 + s;
    int ntile = t >> 2, otile = t & 3;
    int o2 = otile * 16 + mr;
    float bv = lin_b[o2];
    v8f acc = yt[s];                   // C init = f32 y tile (+ bias per column)
    for (int v = 0; v < 8; ++v) acc[v] += bv;
    for (int k0 = 0; k0 < 64; k0 += 32) {
      int n = ntile * 16 + mr;
      V16H a;                          // A rows = n, K = o (contiguous in yL row)
      a.h[0] = *(const v8h*)(&yL[n][k0 + g * 8]);
      a.h[1] = *(const v8h*)(&yL[n][k0 + 16 + g * 8]);
      v16h bb = load_bT16(LWT, o2, 64, k0, g);
      acc = WMMA_F16(a.v, bb, acc);
    }
    for (int v = 0; v < 8; ++v) {
      int n = ntile * 16 + v + 8 * g;
      float r = acc[v];
      op[(size_t)n * 64 + o2] = (r > 0.0f) ? r : 0.0f;
    }
  }
}

extern "C" void kernel_launch(void* const* d_in, const int* in_sizes, int n_in,
                              void* d_out, int out_size, void* d_ws, size_t ws_size,
                              hipStream_t stream) {
  const float* x     = (const float*)d_in[0];   // [8,256,256,64]
  const float* w1    = (const float*)d_in[1];   // [64,64,16,16,2]
  const float* w2    = (const float*)d_in[2];   // [64,64,16,16,2]
  const float* lin_w = (const float*)d_in[3];   // [64,64]
  const float* lin_b = (const float*)d_in[4];   // [64]
  float* out = (float*)d_out;                   // [8,256,256,64]
  char* ws = (char*)d_ws;

  _Float16* TA  = (_Float16*)(ws + OFF_TA);
  _Float16* A2  = (_Float16*)(ws + OFF_A2);
  _Float16* A4  = (_Float16*)(ws + OFF_A4);
  _Float16* A5  = (_Float16*)(ws + OFF_A5);
  _Float16* LWT = (_Float16*)(ws + OFF_LWT);
  _Float16* WBT = (_Float16*)(ws + OFF_WBT);
  _Float16* X1  = (_Float16*)(ws + OFF_X1);
  _Float16* X2  = (_Float16*)(ws + OFF_X2);
  _Float16* OFT = (_Float16*)(ws + OFF_OFT);
  _Float16* Y1  = (_Float16*)(ws + OFF_Y1);

  k_tables<<<128, 256, 0, stream>>>(TA, A2, A4, A5, LWT, lin_w);
  k_wbt<<<32768, 256, 0, stream>>>(w1, w2, WBT);
  k_stageA<<<2048, 256, 0, stream>>>(x, TA, X1);
  k_stageB<<<128, 256, 0, stream>>>(X1, A2, X2);
  k_stageC<<<512, 256, 0, stream>>>(X2, WBT, OFT);
  k_stageD<<<128, 256, 0, stream>>>(OFT, A4, Y1);
  k_stageE<<<2048, 256, 0, stream>>>(Y1, A5, LWT, lin_b, out);
}
